// Attention_45320494907969
// MI455X (gfx1250) — matrix operand order repaired and checked by reading, not compiled, not run on hardware
//
#include <hip/hip_runtime.h>
#include <hip/hip_bf16.h>

// ---------------------------------------------------------------------------
// Shapes (fixed by the reference)
// ---------------------------------------------------------------------------
#define HEADS    16
#define DIMS_    2048
#define HEAD_DIM 128
#define BATCH_   128
#define SEQ_     128
#define MTOT     (BATCH_ * SEQ_)      // 16384 rows for the big GEMMs

// GEMM tiling
#define BM 128
#define BN 128
#define BK 32
#define LDSS 40                       // 32 + 8 pad, in bf16 units (80B row)

typedef __attribute__((ext_vector_type(16))) __bf16 v16bf;
typedef __attribute__((ext_vector_type(8)))  float  v8f;
typedef unsigned int u32x4 __attribute__((ext_vector_type(4)));
typedef int          i32x4 __attribute__((ext_vector_type(4)));
typedef int          i32x8 __attribute__((ext_vector_type(8)));

#if defined(__has_builtin)
#if __has_builtin(__builtin_amdgcn_tensor_load_to_lds)
#define HAS_TDM 1
#endif
#endif
#ifndef HAS_TDM
#define HAS_TDM 0
#endif

// ---------------------------------------------------------------------------
// float -> bf16 (round to nearest even), as raw u16
// ---------------------------------------------------------------------------
__device__ __forceinline__ unsigned short f2bfu(float f) {
    union { float f; unsigned u; } v; v.f = f;
    unsigned r = (v.u + 0x7FFFu + ((v.u >> 16) & 1u)) >> 16;
    return (unsigned short)r;
}
__device__ __forceinline__ unsigned pack2(float a, float b) {
    return (unsigned)f2bfu(a) | ((unsigned)f2bfu(b) << 16);
}

// ---------------------------------------------------------------------------
// TDM: DMA a 2D tile (tile_d1 rows x tile_d0 bf16 elems) from global memory
// (row pitch stride0 elems) into LDS at byte offset lds_addr, inserting 16B
// of padding after every 64B row chunk => LDS row stride 80B (= LDSS shorts).
// Issued by one wave; caller must s_wait_tensorcnt + barrier before use.
// ---------------------------------------------------------------------------
#if HAS_TDM
__device__ __forceinline__ void tdm_load_tile_bf16(
    const void* gptr, unsigned lds_addr,
    unsigned tensor_d0, unsigned tensor_d1, unsigned stride0,
    unsigned tile_d0, unsigned tile_d1)
{
    unsigned long long ga = (unsigned long long)(uintptr_t)gptr;
    u32x4 g0;
    g0.x = 1u;                                    // count=1 (valid), user mode
    g0.y = lds_addr;                              // LDS byte address
    g0.z = (unsigned)(ga & 0xFFFFFFFFu);          // global addr [31:0]
    g0.w = (unsigned)((ga >> 32) & 0x1FFFFFFu)    // global addr [56:32]
         | (2u << 30);                            // type = 2 ("image")
    i32x8 g1;
    g1[0] = (1 << 16)                             // data_size = 2 bytes
          | (1 << 20)                             // pad_enable
          | (3 << 22)                             // pad_interval: 16 DWORDs
          | (3 << 25);                            // pad_amount:   4 DWORDs
    g1[1] = (int)((tensor_d0 & 0xFFFFu) << 16);                   // dim0 lo16
    g1[2] = (int)(((tensor_d0 >> 16) & 0xFFFFu)
                | ((tensor_d1 & 0xFFFFu) << 16));                 // dim0 hi / dim1 lo
    g1[3] = (int)(((tensor_d1 >> 16) & 0xFFFFu)
                | ((tile_d0 & 0xFFFFu) << 16));                   // dim1 hi / tile0
    g1[4] = (int)(tile_d1 & 0xFFFFu);                             // tile1, tile2=0
    g1[5] = (int)stride0;                                         // dim0 stride lo32
    g1[6] = 0;                                                    // stride hi / dim1 stride
    g1[7] = 0;
    i32x4 z4 = {0, 0, 0, 0};
#if __clang_major__ >= 23
    i32x8 z8 = {0, 0, 0, 0, 0, 0, 0, 0};
    __builtin_amdgcn_tensor_load_to_lds(g0, g1, z4, z4, z8, 0);
#else
    __builtin_amdgcn_tensor_load_to_lds(g0, g1, z4, z4, 0);
#endif
}
#endif

// ---------------------------------------------------------------------------
// WMMA fragment loaders from LDS (bf16, 16x16x32)
// ---------------------------------------------------------------------------
__device__ __forceinline__ v16bf frag_a(const unsigned short* s, int lane) {
    const unsigned short* p = s + (lane & 15) * LDSS + ((lane >> 4) << 3);
    union { uint4 q[2]; v16bf v; } u;
    u.q[0] = *(const uint4*)(p);
    u.q[1] = *(const uint4*)(p + 16);
    return u.v;
}
__device__ __forceinline__ v16bf frag_b(const unsigned short* s, int lane) {
    const unsigned short* p = s + (lane & 15) * LDSS + ((lane >> 4) << 3);
    union { uint4 q[2]; v16bf v; } u;
    u.q[0] = *(const uint4*)(p);
    u.q[1] = *(const uint4*)(p + 16);
    return u.v;
}

// ---------------------------------------------------------------------------
// GEMM:  Y[M,N] = X[M,K] @ W[N,K]^T + bias[N]
// Block 256 threads = 8 waves in a 4(M) x 2(N) grid; wave tile 32x64.
// Double-buffered LDS: stage chunk k+1 (TDM + global loads) during compute k.
// ---------------------------------------------------------------------------
template<bool X_F32, bool OUT_F32>
__global__ __launch_bounds__(256)
void gemm_xwT_bias(const void* __restrict__ Xv, const float* __restrict__ W,
                   const float* __restrict__ bias, void* __restrict__ Yv,
                   int M, int N, int K)
{
    __shared__ unsigned short As[2][BM * LDSS];
    __shared__ unsigned short Bs[2][BN * LDSS];

    const int tid  = threadIdx.x;
    const int lane = tid & 31;
    const int wave = tid >> 5;
    const int wr   = wave >> 1;          // 0..3 : M sub-block of 32 rows
    const int wc   = wave & 1;           // 0..1 : N sub-block of 64 cols
    const int mBlk = (int)blockIdx.y * BM;
    const int nBlk = (int)blockIdx.x * BN;

    const int ldRow = tid >> 1;          // 0..127
    const int ldCol = (tid & 1) << 4;    // 0 or 16

    v8f acc[2][4] = {};

    auto stage = [&](int kt, int buf) {
        // ---- X tile -> As[buf] ----
        if constexpr (X_F32) {
            const float* src = (const float*)Xv + (size_t)(mBlk + ldRow) * K + kt + ldCol;
            unsigned* dst = (unsigned*)(As[buf] + ldRow * LDSS + ldCol);
#pragma unroll
            for (int i = 0; i < 4; ++i) {
                float4 f = ((const float4*)src)[i];
                dst[i * 2 + 0] = pack2(f.x, f.y);
                dst[i * 2 + 1] = pack2(f.z, f.w);
            }
        } else {
#if HAS_TDM
            if (wave == 0)
                tdm_load_tile_bf16((const unsigned short*)Xv + (size_t)mBlk * K + kt,
                                   (unsigned)(uintptr_t)(void*)&As[buf][0],
                                   (unsigned)K, (unsigned)M, (unsigned)K, BK, BM);
#else
            const uint4* src = (const uint4*)((const unsigned short*)Xv +
                                              (size_t)(mBlk + ldRow) * K + kt + ldCol);
            uint4* dst = (uint4*)(As[buf] + ldRow * LDSS + ldCol);
            dst[0] = src[0];
            dst[1] = src[1];
#endif
        }
        // ---- W tile -> Bs[buf] (n-major, f32 -> bf16) ----
        {
            const float* src = W + (size_t)(nBlk + ldRow) * K + kt + ldCol;
            unsigned* dst = (unsigned*)(Bs[buf] + ldRow * LDSS + ldCol);
#pragma unroll
            for (int i = 0; i < 4; ++i) {
                float4 f = ((const float4*)src)[i];
                dst[i * 2 + 0] = pack2(f.x, f.y);
                dst[i * 2 + 1] = pack2(f.z, f.w);
            }
        }
    };
    auto sync_stage = [&]() {
#if HAS_TDM
        if constexpr (!X_F32) {
            if (wave == 0) __builtin_amdgcn_s_wait_tensorcnt(0);
        }
#endif
        __syncthreads();
    };
    auto compute = [&](int buf) {
        v16bf a0 = frag_a(As[buf] + (wr * 32 +  0) * LDSS, lane);
        v16bf a1 = frag_a(As[buf] + (wr * 32 + 16) * LDSS, lane);
#pragma unroll
        for (int j = 0; j < 4; ++j) {
            v16bf b = frag_b(Bs[buf] + (wc * 64 + j * 16) * LDSS, lane);
            acc[0][j] = __builtin_amdgcn_wmma_f32_16x16x32_bf16(
                false, a0, false, b, (short)0, acc[0][j], false, false);
            acc[1][j] = __builtin_amdgcn_wmma_f32_16x16x32_bf16(
                false, a1, false, b, (short)0, acc[1][j], false, false);
        }
    };

    const int NK = K / BK;
    stage(0, 0);
    sync_stage();
    for (int k = 0; k < NK; ++k) {
        const int cur = k & 1;
        if (k + 1 < NK) stage((k + 1) * BK, 1 - cur);   // overlap with compute
        compute(cur);
        if (k + 1 < NK) sync_stage();
    }

    // ---- epilogue: +bias, store ----
    const int nc = lane & 15;
    const int mo = (lane >> 4) << 3;
#pragma unroll
    for (int i = 0; i < 2; ++i) {
        const int m0 = mBlk + wr * 32 + i * 16 + mo;
#pragma unroll
        for (int j = 0; j < 4; ++j) {
            const int col = nBlk + wc * 64 + j * 16 + nc;
            const float bv = bias[col];
#pragma unroll
            for (int r = 0; r < 8; ++r) {
                const float val = acc[i][j][r] + bv;
                const size_t off = (size_t)(m0 + r) * N + col;
                if (OUT_F32) ((float*)Yv)[off] = val;
                else         ((unsigned short*)Yv)[off] = f2bfu(val);
            }
        }
    }
}

// ---------------------------------------------------------------------------
// qk_kernel: one block per (b,h).
//   qk[i,j] = (sum_d Q[i,d] * K[127-d, 127-j]) / sqrt(128), masked -> 1e-9
// ---------------------------------------------------------------------------
__global__ __launch_bounds__(256)
void qk_kernel(const unsigned short* __restrict__ Qb,
               const unsigned short* __restrict__ Kb,
               const unsigned char*  __restrict__ mask,
               float* __restrict__ qk)
{
    __shared__ unsigned short As[2][SEQ_ * LDSS];
    __shared__ unsigned short Bs[2][SEQ_ * LDSS];

    const int tid  = threadIdx.x;
    const int lane = tid & 31;
    const int wave = tid >> 5;
    const int wr   = wave >> 1;
    const int wc   = wave & 1;
    const size_t base = (size_t)blockIdx.x * SEQ_ * SEQ_;
    const unsigned short* Q  = Qb + base;
    const unsigned short* Kk = Kb + base;

    v8f acc[2][4] = {};

    auto stage = [&](int kt, int buf) {
        // A: Q rows (TDM when available)
#if HAS_TDM
        if (wave == 0)
            tdm_load_tile_bf16(Q + kt, (unsigned)(uintptr_t)(void*)&As[buf][0],
                               SEQ_, SEQ_, SEQ_, BK, SEQ_);
#else
        {
            const int ldRow = tid >> 1, ldCol = (tid & 1) << 4;
            const uint4* src = (const uint4*)(Q + ldRow * SEQ_ + kt + ldCol);
            uint4* dst = (uint4*)(As[buf] + ldRow * LDSS + ldCol);
            dst[0] = src[0];
            dst[1] = src[1];
        }
#endif
        // B: Bs[j][d-kt] = K[127-d][127-j]  (flip both dims + transpose)
        {
            const int rr = tid >> 3;            // d = kt + rr
            const int r  = 127 - kt - rr;       // source row
            const int c0 = (tid & 7) << 4;
            const unsigned short* src = Kk + r * SEQ_ + c0;
#pragma unroll
            for (int i = 0; i < 16; ++i)
                Bs[buf][(127 - (c0 + i)) * LDSS + rr] = src[i];
        }
    };
    auto sync_stage = [&]() {
#if HAS_TDM
        if (wave == 0) __builtin_amdgcn_s_wait_tensorcnt(0);
#endif
        __syncthreads();
    };
    auto compute = [&](int buf) {
        v16bf a0 = frag_a(As[buf] + (wr * 32 +  0) * LDSS, lane);
        v16bf a1 = frag_a(As[buf] + (wr * 32 + 16) * LDSS, lane);
#pragma unroll
        for (int j = 0; j < 4; ++j) {
            v16bf b = frag_b(Bs[buf] + (wc * 64 + j * 16) * LDSS, lane);
            acc[0][j] = __builtin_amdgcn_wmma_f32_16x16x32_bf16(
                false, a0, false, b, (short)0, acc[0][j], false, false);
            acc[1][j] = __builtin_amdgcn_wmma_f32_16x16x32_bf16(
                false, a1, false, b, (short)0, acc[1][j], false, false);
        }
    };

    const int NK = SEQ_ / BK;   // 4
    stage(0, 0);
    sync_stage();
    for (int k = 0; k < NK; ++k) {
        const int cur = k & 1;
        if (k + 1 < NK) stage((k + 1) * BK, 1 - cur);
        compute(cur);
        if (k + 1 < NK) sync_stage();
    }

    const float scale = 0.088388347648318447f;   // 1/sqrt(128)
    const int nc = lane & 15;
    const int mo = (lane >> 4) << 3;
#pragma unroll
    for (int i = 0; i < 2; ++i) {
        const int m0 = wr * 32 + i * 16 + mo;
#pragma unroll
        for (int j = 0; j < 4; ++j) {
            const int col = wc * 64 + j * 16 + nc;
#pragma unroll
            for (int r = 0; r < 8; ++r) {
                const size_t off = base + (size_t)(m0 + r) * SEQ_ + col;
                const float val = acc[i][j][r] * scale;
                qk[off] = mask[off] ? val : 1e-9f;
            }
        }
    }
}

// ---------------------------------------------------------------------------
// softmax over the HEADS axis
// ---------------------------------------------------------------------------
__global__ __launch_bounds__(256)
void softmax_heads(const float* __restrict__ qk, unsigned short* __restrict__ wgt)
{
    const size_t idx = (size_t)blockIdx.x * 256 + threadIdx.x;
    const int b  = (int)(idx >> 14);              // S*S = 16384
    const int ij = (int)(idx & 16383);
    const size_t base = ((size_t)b << 18) + ij;   // H*S*S per batch

    float v[HEADS];
    float mx = -3.4e38f;
#pragma unroll
    for (int h = 0; h < HEADS; ++h) {
        v[h] = qk[base + (size_t)h * 16384];
        mx = fmaxf(mx, v[h]);
    }
    float s = 0.f;
#pragma unroll
    for (int h = 0; h < HEADS; ++h) {
        v[h] = __expf(v[h] - mx);
        s += v[h];
    }
    const float inv = 1.f / s;
#pragma unroll
    for (int h = 0; h < HEADS; ++h)
        wgt[base + (size_t)h * 16384] = f2bfu(v[h] * inv);
}

// ---------------------------------------------------------------------------
// av_kernel: one block per (b,h).  out[i,j] = sum_d V[i,d] * Wgt[d,j]
// ---------------------------------------------------------------------------
__global__ __launch_bounds__(256)
void av_kernel(const unsigned short* __restrict__ Vb,
               const unsigned short* __restrict__ Wg,
               unsigned short* __restrict__ attn)
{
    __shared__ unsigned short As[2][SEQ_ * LDSS];
    __shared__ unsigned short Bs[2][SEQ_ * LDSS];

    const int tid  = threadIdx.x;
    const int lane = tid & 31;
    const int wave = tid >> 5;
    const int wr   = wave >> 1;
    const int wc   = wave & 1;
    const size_t base = (size_t)blockIdx.x * SEQ_ * SEQ_;
    const unsigned short* V = Vb + base;
    const unsigned short* G = Wg + base;

    v8f acc[2][4] = {};

    auto stage = [&](int kt, int buf) {
#if HAS_TDM
        if (wave == 0)
            tdm_load_tile_bf16(V + kt, (unsigned)(uintptr_t)(void*)&As[buf][0],
                               SEQ_, SEQ_, SEQ_, BK, SEQ_);
#else
        {
            const int ldRow = tid >> 1, ldCol = (tid & 1) << 4;
            const uint4* src = (const uint4*)(V + ldRow * SEQ_ + kt + ldCol);
            uint4* dst = (uint4*)(As[buf] + ldRow * LDSS + ldCol);
            dst[0] = src[0];
            dst[1] = src[1];
        }
#endif
        // B: Bs[j][d-kt] = Wgt[d][j]  (transpose staging)
        {
            const int rr = tid >> 3;            // d = kt + rr
            const int c0 = (tid & 7) << 4;
            const unsigned short* src = G + (kt + rr) * SEQ_ + c0;
#pragma unroll
            for (int i = 0; i < 16; ++i)
                Bs[buf][(c0 + i) * LDSS + rr] = src[i];
        }
    };
    auto sync_stage = [&]() {
#if HAS_TDM
        if (wave == 0) __builtin_amdgcn_s_wait_tensorcnt(0);
#endif
        __syncthreads();
    };
    auto compute = [&](int buf) {
        v16bf a0 = frag_a(As[buf] + (wr * 32 +  0) * LDSS, lane);
        v16bf a1 = frag_a(As[buf] + (wr * 32 + 16) * LDSS, lane);
#pragma unroll
        for (int j = 0; j < 4; ++j) {
            v16bf b = frag_b(Bs[buf] + (wc * 64 + j * 16) * LDSS, lane);
            acc[0][j] = __builtin_amdgcn_wmma_f32_16x16x32_bf16(
                false, a0, false, b, (short)0, acc[0][j], false, false);
            acc[1][j] = __builtin_amdgcn_wmma_f32_16x16x32_bf16(
                false, a1, false, b, (short)0, acc[1][j], false, false);
        }
    };

    const int NK = SEQ_ / BK;   // 4
    stage(0, 0);
    sync_stage();
    for (int k = 0; k < NK; ++k) {
        const int cur = k & 1;
        if (k + 1 < NK) stage((k + 1) * BK, 1 - cur);
        compute(cur);
        if (k + 1 < NK) sync_stage();
    }

    const int nc = lane & 15;
    const int mo = (lane >> 4) << 3;
#pragma unroll
    for (int i = 0; i < 2; ++i) {
        const int m0 = wr * 32 + i * 16 + mo;
#pragma unroll
        for (int j = 0; j < 4; ++j) {
            const int col = wc * 64 + j * 16 + nc;
#pragma unroll
            for (int r = 0; r < 8; ++r)
                attn[base + (size_t)(m0 + r) * SEQ_ + col] = f2bfu(acc[i][j][r]);
        }
    }
}

// ---------------------------------------------------------------------------
// Host-side launch
// ---------------------------------------------------------------------------
extern "C" void kernel_launch(void* const* d_in, const int* in_sizes, int n_in,
                              void* d_out, int out_size, void* d_ws, size_t ws_size,
                              hipStream_t stream)
{
    const float* pq  = (const float*)d_in[0];
    const float* pk  = (const float*)d_in[1];
    const float* pv  = (const float*)d_in[2];
    const float* Wq  = (const float*)d_in[3];
    const float* bq  = (const float*)d_in[4];
    const float* Wk  = (const float*)d_in[5];
    const float* bk  = (const float*)d_in[6];
    const float* Wv  = (const float*)d_in[7];
    const float* bv  = (const float*)d_in[8];
    const float* Wfc = (const float*)d_in[9];
    const float* bfc = (const float*)d_in[10];
    const unsigned char* mask = (const unsigned char*)d_in[11];
    float* out = (float*)d_out;

    // workspace: [q_bf 64MB][k_bf 64MB][v_bf 64MB][qk 128MB]
    // wgt aliases q_bf (dead after qk_kernel), attn aliases k_bf.
    unsigned short* q_bf = (unsigned short*)d_ws;
    unsigned short* k_bf = q_bf + (size_t)MTOT * DIMS_;
    unsigned short* v_bf = k_bf + (size_t)MTOT * DIMS_;
    float*          qk   = (float*)(v_bf + (size_t)MTOT * DIMS_);
    unsigned short* wgt  = q_bf;
    unsigned short* attn = k_bf;

    dim3 blk(256);
    dim3 gproj(DIMS_ / BN, MTOT / BM);        // (16, 128)

    gemm_xwT_bias<true,  false><<<gproj, blk, 0, stream>>>(pq, Wq, bq, q_bf, MTOT, DIMS_, DIMS_);
    gemm_xwT_bias<true,  false><<<gproj, blk, 0, stream>>>(pk, Wk, bk, k_bf, MTOT, DIMS_, DIMS_);
    gemm_xwT_bias<true,  false><<<gproj, blk, 0, stream>>>(pv, Wv, bv, v_bf, MTOT, DIMS_, DIMS_);

    qk_kernel<<<BATCH_ * HEADS, blk, 0, stream>>>(q_bf, k_bf, mask, qk);

    softmax_heads<<<(BATCH_ * SEQ_ * SEQ_) / 256, blk, 0, stream>>>(qk, wgt);

    av_kernel<<<BATCH_ * HEADS, blk, 0, stream>>>(v_bf, wgt, attn);

    gemm_xwT_bias<false, true><<<gproj, blk, 0, stream>>>(attn, Wfc, bfc, out, MTOT, DIMS_, DIMS_);
}
